// Prototype_62577673503172
// MI455X (gfx1250) — compile-verified, matrix operand order checked by reading, not run on hardware
//
#include <hip/hip_runtime.h>
#include <hip/hip_bf16.h>

// Problem sizes
#define MB       8192            // batch rows
#define KD       2048            // feature dim
#define NPROTO   3450            // prototypes
#define NP       3456            // padded to 27*128 for WMMA tiling
#define NCLS     345             // classes
#define NC       352             // padded to 11*32
#define EPS      0.05f
#define INV_EPS  20.0f
#define SINK_ITERS 50
#define GSPLIT   64              // row-chunks for the g (column) pass
#define GROWS    (MB / GSPLIT)   // 128 rows per chunk
#define LOG_A   (-9.010913347279289f)   // log(1/8192)
#define LOG_B   (-8.146129510469392f)   // log(1/3450)
#define LOSS_SCALE (0.1f / 6.0f)        // LAMBDA_OT / N_DOMAINS

typedef __attribute__((ext_vector_type(16))) __bf16 v16bf;
typedef __attribute__((ext_vector_type(8)))  float  v8f;

// ---------------------------------------------------------------------------
// WMMA fragment load: 16x32 bf16 tile, one matrix row per lane (l%16),
// K halves split per ISA: h=0 -> K[0..7],K[16..23]; h=1 -> K[8..15],K[24..31].
// lanebase must already include row*ld + 8*h.
// ---------------------------------------------------------------------------
__device__ __forceinline__ v16bf load_frag(const __bf16* lanebase, int k) {
    union { v16bf v; uint4 q[2]; } u;
    u.q[0] = *reinterpret_cast<const uint4*>(lanebase + k);
    u.q[1] = *reinterpret_cast<const uint4*>(lanebase + k + 16);
    return u.v;
}

// Streaming logsumexp helpers
__device__ __forceinline__ void lse_push(float& m, float& s, float x) {
    if (x > m) { s = s * __expf(m - x) + 1.0f; m = x; }
    else       { s += __expf(x - m); }
}
__device__ __forceinline__ void lse_merge(float& m, float& s, float m2, float s2) {
    if (m2 == -__builtin_inff()) return;
    if (m2 > m) { s = s * __expf(m - m2) + s2; m = m2; }
    else        { s += s2 * __expf(m2 - m); }
}

// ---------------------------------------------------------------------------
// Row-wise fp32 -> bf16 convert + inverse L2 norm (rows >= validRows zeroed)
// ---------------------------------------------------------------------------
__global__ __launch_bounds__(256) void norm_convert_k(const float* __restrict__ X,
                                                      __bf16* __restrict__ Xb,
                                                      float* __restrict__ invn,
                                                      int validRows) {
    int row = blockIdx.x;
    int tid = threadIdx.x;
    const float* x = X + (size_t)row * KD;
    __bf16* xb = Xb + (size_t)row * KD;
    if (row >= validRows) {
        for (int d = tid; d < KD; d += 256) xb[d] = (__bf16)0.0f;
        if (tid == 0) invn[row] = 1.0f;
        return;
    }
    float ss = 0.0f;
    for (int d = tid; d < KD; d += 256) {
        float v = x[d];
        xb[d] = (__bf16)v;
        ss += v * v;
    }
    __shared__ float red[256];
    red[tid] = ss; __syncthreads();
    for (int o = 128; o > 0; o >>= 1) {
        if (tid < o) red[tid] += red[tid + o];
        __syncthreads();
    }
    if (tid == 0) invn[row] = 1.0f / fmaxf(sqrtf(red[0]), 1e-12f);
}

__global__ __launch_bounds__(256) void convert_rows_k(const float* __restrict__ X,
                                                      __bf16* __restrict__ Xb,
                                                      int validRows) {
    int row = blockIdx.x;
    int tid = threadIdx.x;
    __bf16* xb = Xb + (size_t)row * KD;
    if (row >= validRows) {
        for (int d = tid; d < KD; d += 256) xb[d] = (__bf16)0.0f;
        return;
    }
    const float* x = X + (size_t)row * KD;
    for (int d = tid; d < KD; d += 256) xb[d] = (__bf16)x[d];
}

// ---------------------------------------------------------------------------
// Main GEMM: G[MB x NP] = featB[MB x KD] * protB[NP x KD]^T  (bf16 -> f32)
// Block 256 threads = 8 waves (2 in M x 4 in N). Wave tile 64x32 (4x2 frags).
// Block tile 128x128. Grid (NP/128=27, MB/128=64). k-loop double-buffered so
// next step's global loads overlap this step's 8 WMMAs.
// ---------------------------------------------------------------------------
__global__ __launch_bounds__(256) void gemm_bf16_k(const __bf16* __restrict__ A,
                                                   const __bf16* __restrict__ Bm,
                                                   float* __restrict__ G) {
    int w    = threadIdx.x >> 5;
    int lane = threadIdx.x & 31;
    int r    = lane & 15;
    int h    = lane >> 4;
    int wm   = w & 1, wn = w >> 1;
    int m0   = blockIdx.y * 128 + wm * 64;
    int n0   = blockIdx.x * 128 + wn * 32;

    const __bf16* abase[4];
    const __bf16* bbase[2];
#pragma unroll
    for (int i = 0; i < 4; ++i)
        abase[i] = A + (size_t)(m0 + 16 * i + r) * KD + 8 * h;
#pragma unroll
    for (int j = 0; j < 2; ++j)
        bbase[j] = Bm + (size_t)(n0 + 16 * j + r) * KD + 8 * h;

    v16bf a0[4], b0[2], a1[4], b1[2];
#pragma unroll
    for (int i = 0; i < 4; ++i) a0[i] = load_frag(abase[i], 0);
#pragma unroll
    for (int j = 0; j < 2; ++j) b0[j] = load_frag(bbase[j], 0);

    v8f acc[4][2] = {};
    for (int k = 0; k < KD; k += 64) {
        // prefetch k+32 (always valid: last k = KD-64)
#pragma unroll
        for (int i = 0; i < 4; ++i) a1[i] = load_frag(abase[i], k + 32);
#pragma unroll
        for (int j = 0; j < 2; ++j) b1[j] = load_frag(bbase[j], k + 32);
        // compute on buffer 0
#pragma unroll
        for (int i = 0; i < 4; ++i)
#pragma unroll
            for (int j = 0; j < 2; ++j)
                acc[i][j] = __builtin_amdgcn_wmma_f32_16x16x32_bf16(
                    false, a0[i], false, b0[j], (short)0, acc[i][j], false, false);
        // prefetch k+64
        if (k + 64 < KD) {
#pragma unroll
            for (int i = 0; i < 4; ++i) a0[i] = load_frag(abase[i], k + 64);
#pragma unroll
            for (int j = 0; j < 2; ++j) b0[j] = load_frag(bbase[j], k + 64);
        }
        // compute on buffer 1
#pragma unroll
        for (int i = 0; i < 4; ++i)
#pragma unroll
            for (int j = 0; j < 2; ++j)
                acc[i][j] = __builtin_amdgcn_wmma_f32_16x16x32_bf16(
                    false, a1[i], false, b1[j], (short)0, acc[i][j], false, false);
    }
    // C/D layout: (vgpr rr, lane) -> m = rr + 8*(lane>>4), n = lane&15
#pragma unroll
    for (int i = 0; i < 4; ++i)
#pragma unroll
        for (int j = 0; j < 2; ++j)
#pragma unroll
            for (int rr = 0; rr < 8; ++rr)
                G[(size_t)(m0 + 16 * i + rr + 8 * h) * NP + n0 + 16 * j + r] = acc[i][j][rr];
}

// ---------------------------------------------------------------------------
// argmin_j C_ij over j<NPROTO, C = 1 - G*invnf_i*invnp_j (first-index ties)
// ---------------------------------------------------------------------------
__global__ __launch_bounds__(256) void argmin_k(const float* __restrict__ G,
                                                const float* __restrict__ invnf,
                                                const float* __restrict__ invnp,
                                                int* __restrict__ nearest,
                                                float* __restrict__ outNearF) {
    int i = blockIdx.x, tid = threadIdx.x;
    float sf = invnf[i];
    float bv = __builtin_inff();
    int   bi = 0x7fffffff;
    const float* g = G + (size_t)i * NP;
    for (int j = tid; j < NPROTO; j += 256) {
        float c = 1.0f - g[j] * sf * invnp[j];
        if (c < bv || (c == bv && j < bi)) { bv = c; bi = j; }
    }
    __shared__ float rv[256];
    __shared__ int   ri[256];
    rv[tid] = bv; ri[tid] = bi; __syncthreads();
    for (int o = 128; o > 0; o >>= 1) {
        if (tid < o) {
            float v2 = rv[tid + o]; int i2 = ri[tid + o];
            if (v2 < rv[tid] || (v2 == rv[tid] && i2 < ri[tid])) { rv[tid] = v2; ri[tid] = i2; }
        }
        __syncthreads();
    }
    if (tid == 0) { nearest[i] = ri[0]; outNearF[i] = (float)ri[0]; }
}

// ---------------------------------------------------------------------------
// Sinkhorn f-pass: f_i = eps*log(a_i) - eps*lse_j((g_j - C_ij)/eps)
// One block per row; float4 loads (G rows 16B aligned, NP=3456).
// ---------------------------------------------------------------------------
__global__ __launch_bounds__(256) void sink_f_k(const float* __restrict__ G,
                                                const float* __restrict__ invnf,
                                                const float* __restrict__ invnp,
                                                const float* __restrict__ gvec,
                                                float* __restrict__ fvec) {
    int i = blockIdx.x, tid = threadIdx.x;
    float sf = invnf[i] * INV_EPS;           // x = (g-1)*INV_EPS + G*sf*invnp
    const float* grow = G + (size_t)i * NP;
    float m = -__builtin_inff(), s = 0.0f;
    for (int j = tid * 4; j + 3 < NPROTO; j += 1024) {
        float4 gq = *reinterpret_cast<const float4*>(grow + j);
        float4 np = *reinterpret_cast<const float4*>(invnp + j);
        float4 gg = *reinterpret_cast<const float4*>(gvec + j);
        lse_push(m, s, (gg.x - 1.0f) * INV_EPS + gq.x * sf * np.x);
        lse_push(m, s, (gg.y - 1.0f) * INV_EPS + gq.y * sf * np.y);
        lse_push(m, s, (gg.z - 1.0f) * INV_EPS + gq.z * sf * np.z);
        lse_push(m, s, (gg.w - 1.0f) * INV_EPS + gq.w * sf * np.w);
    }
    for (int j = (NPROTO & ~3) + tid; j < NPROTO; j += 256)   // tail 3448,3449
        lse_push(m, s, (gvec[j] - 1.0f) * INV_EPS + grow[j] * sf * invnp[j]);
    __shared__ float sm[256], ssh[256];
    sm[tid] = m; ssh[tid] = s; __syncthreads();
    for (int o = 128; o > 0; o >>= 1) {
        if (tid < o) lse_merge(sm[tid], ssh[tid], sm[tid + o], ssh[tid + o]);
        __syncthreads();
    }
    if (tid == 0) fvec[i] = EPS * LOG_A - EPS * (sm[0] + __logf(ssh[0]));
}

// g column pass: each thread owns 4 consecutive columns (float4 loads),
// split across GSPLIT row chunks. Grid ((3450+1023)/1024, GSPLIT).
__global__ __launch_bounds__(256) void sink_g_partial_k(const float* __restrict__ G,
                                                        const float* __restrict__ invnf,
                                                        const float* __restrict__ invnp,
                                                        const float* __restrict__ fvec,
                                                        float* __restrict__ pm,
                                                        float* __restrict__ ps) {
    int j0 = (blockIdx.x * 256 + threadIdx.x) * 4;
    if (j0 >= NPROTO) return;
    int chunk = blockIdx.y;
    int i0 = chunk * GROWS, i1 = i0 + GROWS;
    float* pmc = pm + (size_t)chunk * NP;
    float* psc = ps + (size_t)chunk * NP;

    if (j0 + 3 < NPROTO) {
        float4 sp = *reinterpret_cast<const float4*>(invnp + j0);
        float m0 = -__builtin_inff(), m1 = m0, m2 = m0, m3 = m0;
        float s0 = 0.f, s1 = 0.f, s2 = 0.f, s3 = 0.f;
        for (int i = i0; i < i1; ++i) {
            float4 gq = *reinterpret_cast<const float4*>(G + (size_t)i * NP + j0);
            float sf = invnf[i] * INV_EPS;
            float c0 = (fvec[i] - 1.0f) * INV_EPS;
            lse_push(m0, s0, c0 + gq.x * sf * sp.x);
            lse_push(m1, s1, c0 + gq.y * sf * sp.y);
            lse_push(m2, s2, c0 + gq.z * sf * sp.z);
            lse_push(m3, s3, c0 + gq.w * sf * sp.w);
        }
        pmc[j0] = m0; pmc[j0 + 1] = m1; pmc[j0 + 2] = m2; pmc[j0 + 3] = m3;
        psc[j0] = s0; psc[j0 + 1] = s1; psc[j0 + 2] = s2; psc[j0 + 3] = s3;
    } else {
        for (int j = j0; j < NPROTO; ++j) {
            float sp = invnp[j];
            float m = -__builtin_inff(), s = 0.0f;
            for (int i = i0; i < i1; ++i) {
                float sf = invnf[i] * INV_EPS;
                float c0 = (fvec[i] - 1.0f) * INV_EPS;
                lse_push(m, s, c0 + G[(size_t)i * NP + j] * sf * sp);
            }
            pmc[j] = m; psc[j] = s;
        }
    }
}

__global__ __launch_bounds__(256) void sink_g_combine_k(const float* __restrict__ pm,
                                                        const float* __restrict__ ps,
                                                        float* __restrict__ gvec) {
    int j = blockIdx.x * 256 + threadIdx.x;
    if (j >= NPROTO) return;
    float m = -__builtin_inff(), s = 0.0f;
    for (int c = 0; c < GSPLIT; ++c)
        lse_merge(m, s, pm[(size_t)c * NP + j], ps[(size_t)c * NP + j]);
    gvec[j] = EPS * LOG_B - EPS * (m + __logf(s));
}

// ot_cost = sum_ij exp((f_i+g_j-C_ij)/eps) * C_ij -> scalars[0]
__global__ __launch_bounds__(256) void ot_cost_k(const float* __restrict__ G,
                                                 const float* __restrict__ invnf,
                                                 const float* __restrict__ invnp,
                                                 const float* __restrict__ fvec,
                                                 const float* __restrict__ gvec,
                                                 float* __restrict__ scalars) {
    int i = blockIdx.x, tid = threadIdx.x;
    float fi = fvec[i], sf = invnf[i];
    const float* grow = G + (size_t)i * NP;
    float acc = 0.0f;
    for (int j = tid * 4; j + 3 < NPROTO; j += 1024) {
        float4 gq = *reinterpret_cast<const float4*>(grow + j);
        float4 np = *reinterpret_cast<const float4*>(invnp + j);
        float4 gg = *reinterpret_cast<const float4*>(gvec + j);
        float c0 = 1.0f - gq.x * sf * np.x;
        float c1 = 1.0f - gq.y * sf * np.y;
        float c2 = 1.0f - gq.z * sf * np.z;
        float c3 = 1.0f - gq.w * sf * np.w;
        acc += __expf((fi + gg.x - c0) * INV_EPS) * c0;
        acc += __expf((fi + gg.y - c1) * INV_EPS) * c1;
        acc += __expf((fi + gg.z - c2) * INV_EPS) * c2;
        acc += __expf((fi + gg.w - c3) * INV_EPS) * c3;
    }
    for (int j = (NPROTO & ~3) + tid; j < NPROTO; j += 256) {
        float c = 1.0f - grow[j] * sf * invnp[j];
        acc += __expf((fi + gvec[j] - c) * INV_EPS) * c;
    }
    __shared__ float red[256];
    red[tid] = acc; __syncthreads();
    for (int o = 128; o > 0; o >>= 1) {
        if (tid < o) red[tid] += red[tid + o];
        __syncthreads();
    }
    if (tid == 0) atomicAdd(&scalars[0], red[0]);
}

// Row LSE of raw G (for softmax + prototype CE); adds CE term to scalars[1]
__global__ __launch_bounds__(256) void lse_proto_k(const float* __restrict__ G,
                                                   const int* __restrict__ nearest,
                                                   float* __restrict__ lse,
                                                   float* __restrict__ scalars) {
    int i = blockIdx.x, tid = threadIdx.x;
    const float* grow = G + (size_t)i * NP;
    float m = -__builtin_inff(), s = 0.0f;
    for (int j = tid * 4; j + 3 < NPROTO; j += 1024) {
        float4 gq = *reinterpret_cast<const float4*>(grow + j);
        lse_push(m, s, gq.x); lse_push(m, s, gq.y);
        lse_push(m, s, gq.z); lse_push(m, s, gq.w);
    }
    for (int j = (NPROTO & ~3) + tid; j < NPROTO; j += 256) lse_push(m, s, grow[j]);
    __shared__ float sm[256], ssh[256];
    sm[tid] = m; ssh[tid] = s; __syncthreads();
    for (int o = 128; o > 0; o >>= 1) {
        if (tid < o) lse_merge(sm[tid], ssh[tid], sm[tid + o], ssh[tid + o]);
        __syncthreads();
    }
    if (tid == 0) {
        float L = sm[0] + __logf(ssh[0]);
        lse[i] = L;
        atomicAdd(&scalars[1], (L - grow[nearest[i]]) * (1.0f / MB));
    }
}

__global__ __launch_bounds__(256) void softmax_k(const float* __restrict__ G,
                                                 const float* __restrict__ lse,
                                                 float* __restrict__ out) {
    int i = blockIdx.x, tid = threadIdx.x;
    float L = lse[i];
    const float* grow = G + (size_t)i * NP;
    float* orow = out + (size_t)i * NPROTO;   // only 4B-aligned rows: scalar stores
    for (int j = tid * 4; j + 3 < NPROTO; j += 1024) {
        float4 gq = *reinterpret_cast<const float4*>(grow + j);
        orow[j]     = __expf(gq.x - L);
        orow[j + 1] = __expf(gq.y - L);
        orow[j + 2] = __expf(gq.z - L);
        orow[j + 3] = __expf(gq.w - L);
    }
    for (int j = (NPROTO & ~3) + tid; j < NPROTO; j += 256)
        orow[j] = __expf(grow[j] - L);
}

// ---------------------------------------------------------------------------
// Quantized logits: Q[MB x NC] = protB[nearest[i]] * Wb^T via WMMA with
// gathered A rows, double-buffered. Grid (NC/32, MB/128), 8 waves in M.
// ---------------------------------------------------------------------------
__global__ __launch_bounds__(256) void gemm_quant_k(const __bf16* __restrict__ P,
                                                    const __bf16* __restrict__ Wb,
                                                    const int* __restrict__ nearest,
                                                    float* __restrict__ Q) {
    int w    = threadIdx.x >> 5;
    int lane = threadIdx.x & 31;
    int r    = lane & 15;
    int h    = lane >> 4;
    int m0   = blockIdx.y * 128 + w * 16;
    int n0   = blockIdx.x * 32;

    const __bf16* abase  = P + (size_t)nearest[m0 + r] * KD + 8 * h;  // gathered row
    const __bf16* bbase0 = Wb + (size_t)(n0 + r) * KD + 8 * h;
    const __bf16* bbase1 = Wb + (size_t)(n0 + 16 + r) * KD + 8 * h;

    v16bf a_0 = load_frag(abase, 0), p_0 = load_frag(bbase0, 0), q_0 = load_frag(bbase1, 0);
    v16bf a_1, p_1, q_1;
    v8f acc[2] = {};
    for (int k = 0; k < KD; k += 64) {
        a_1 = load_frag(abase, k + 32);
        p_1 = load_frag(bbase0, k + 32);
        q_1 = load_frag(bbase1, k + 32);
        acc[0] = __builtin_amdgcn_wmma_f32_16x16x32_bf16(false, a_0, false, p_0, (short)0, acc[0], false, false);
        acc[1] = __builtin_amdgcn_wmma_f32_16x16x32_bf16(false, a_0, false, q_0, (short)0, acc[1], false, false);
        if (k + 64 < KD) {
            a_0 = load_frag(abase, k + 64);
            p_0 = load_frag(bbase0, k + 64);
            q_0 = load_frag(bbase1, k + 64);
        }
        acc[0] = __builtin_amdgcn_wmma_f32_16x16x32_bf16(false, a_1, false, p_1, (short)0, acc[0], false, false);
        acc[1] = __builtin_amdgcn_wmma_f32_16x16x32_bf16(false, a_1, false, q_1, (short)0, acc[1], false, false);
    }
#pragma unroll
    for (int j = 0; j < 2; ++j)
#pragma unroll
        for (int rr = 0; rr < 8; ++rr)
            Q[(size_t)(m0 + rr + 8 * h) * NC + n0 + 16 * j + r] = acc[j][rr];
}

// quantized CE over 345 classes -> scalars[2]
__global__ __launch_bounds__(256) void quant_ce_k(const float* __restrict__ Q,
                                                  const int* __restrict__ labels,
                                                  float* __restrict__ scalars) {
    int i = blockIdx.x, tid = threadIdx.x;
    const float* qrow = Q + (size_t)i * NC;
    float m = -__builtin_inff(), s = 0.0f;
    for (int j = tid; j < NCLS; j += 256) lse_push(m, s, qrow[j]);
    __shared__ float sm[256], ssh[256];
    sm[tid] = m; ssh[tid] = s; __syncthreads();
    for (int o = 128; o > 0; o >>= 1) {
        if (tid < o) lse_merge(sm[tid], ssh[tid], sm[tid + o], ssh[tid + o]);
        __syncthreads();
    }
    if (tid == 0) {
        float L = sm[0] + __logf(ssh[0]);
        atomicAdd(&scalars[2], (L - qrow[labels[i]]) * (1.0f / MB));
    }
}

__global__ __launch_bounds__(256) void init_k(float* __restrict__ fvec,
                                              float* __restrict__ gvec,
                                              float* __restrict__ scalars) {
    int t = blockIdx.x * 256 + threadIdx.x;
    if (t < MB) fvec[t] = 0.0f;
    if (t < NP) gvec[t] = 0.0f;
    if (t < 4)  scalars[t] = 0.0f;
}

__global__ void finalize_k(const float* __restrict__ scalars, float* __restrict__ out0) {
    if (threadIdx.x == 0)
        out0[0] = scalars[0] + LOSS_SCALE * scalars[1] + LOSS_SCALE * scalars[2];
}

// ---------------------------------------------------------------------------
extern "C" void kernel_launch(void* const* d_in, const int* in_sizes, int n_in,
                              void* d_out, int out_size, void* d_ws, size_t ws_size,
                              hipStream_t stream) {
    (void)in_sizes; (void)n_in; (void)out_size; (void)ws_size;
    const float* features  = (const float*)d_in[0];   // [8192, 2048]
    const int*   labels    = (const int*)  d_in[1];   // [8192]
    const float* weight    = (const float*)d_in[2];   // [345, 2048]
    const float* prototype = (const float*)d_in[3];   // [3450, 2048]
    float* out = (float*)d_out;                       // [1 + 8192 + 8192*3450]

    char* ws = (char*)d_ws;
    size_t off = 0;
    auto alloc = [&](size_t bytes) { size_t r = off; off += (bytes + 255) & ~(size_t)255; return r; };
    float*  G      = (float*) (ws + alloc((size_t)MB * NP * 4));
    __bf16* featB  = (__bf16*)(ws + alloc((size_t)MB * KD * 2));
    __bf16* protB  = (__bf16*)(ws + alloc((size_t)NP * KD * 2));
    __bf16* wB     = (__bf16*)(ws + alloc((size_t)NC * KD * 2));
    float*  invnf  = (float*) (ws + alloc((size_t)MB * 4));
    float*  invnp  = (float*) (ws + alloc((size_t)NP * 4));
    float*  fvec   = (float*) (ws + alloc((size_t)MB * 4));
    float*  gvec   = (float*) (ws + alloc((size_t)NP * 4));
    float*  lse    = (float*) (ws + alloc((size_t)MB * 4));
    int*    nearest= (int*)   (ws + alloc((size_t)MB * 4));
    float*  Q      = (float*) (ws + alloc((size_t)MB * NC * 4));
    float*  pm     = (float*) (ws + alloc((size_t)GSPLIT * NP * 4));
    float*  ps     = (float*) (ws + alloc((size_t)GSPLIT * NP * 4));
    float*  scal   = (float*) (ws + alloc(4 * 4));

    // Prep: converts + norms + init
    norm_convert_k<<<MB, 256, 0, stream>>>(features, featB, invnf, MB);
    norm_convert_k<<<NP, 256, 0, stream>>>(prototype, protB, invnp, NPROTO);
    convert_rows_k<<<NC, 256, 0, stream>>>(weight, wB, NCLS);
    init_k<<<32, 256, 0, stream>>>(fvec, gvec, scal);

    // Big GEMM: G = F * P^T (WMMA bf16, double-buffered)
    gemm_bf16_k<<<dim3(NP / 128, MB / 128), 256, 0, stream>>>(featB, protB, G);

    // Nearest prototype (argmin of cost)
    argmin_k<<<MB, 256, 0, stream>>>(G, invnf, invnp, nearest, out + 1);

    // Sinkhorn (log-domain, 50 iters)
    int gqblocks = (NPROTO + 1023) / 1024;           // quad-column blocks
    int gcblocks = (NPROTO + 255) / 256;
    for (int it = 0; it < SINK_ITERS; ++it) {
        sink_f_k<<<MB, 256, 0, stream>>>(G, invnf, invnp, gvec, fvec);
        sink_g_partial_k<<<dim3(gqblocks, GSPLIT), 256, 0, stream>>>(G, invnf, invnp, fvec, pm, ps);
        sink_g_combine_k<<<gcblocks, 256, 0, stream>>>(pm, ps, gvec);
    }
    ot_cost_k<<<MB, 256, 0, stream>>>(G, invnf, invnp, fvec, gvec, scal);

    // Row LSE of G -> prototype CE + softmax output
    lse_proto_k<<<MB, 256, 0, stream>>>(G, nearest, lse, scal);
    softmax_k<<<MB, 256, 0, stream>>>(G, lse, out + 1 + MB);

    // Quantized logits (gathered-A WMMA GEMM) + CE
    gemm_quant_k<<<dim3(NC / 32, MB / 128), 256, 0, stream>>>(protB, wB, nearest, Q);
    quant_ce_k<<<MB, 256, 0, stream>>>(Q, labels, scal);

    finalize_k<<<1, 1, 0, stream>>>(scal, out);
}
